// VITEncoder_29368986370115
// MI455X (gfx1250) — matrix operand order repaired and checked by reading, not compiled, not run on hardware
//
#include <hip/hip_runtime.h>
#include <hip/hip_bf16.h>
#include <math.h>

// ---------------- problem constants ----------------
#define T_STEPC     16
#define NUM_PATCH   49
#define SEQ         (T_STEPC * NUM_PATCH)   // 784
#define SEQPAD      800                     // 784 padded to 32-multiple (25 chunks)
#define DIMC        512
#define MLP_DIMC    2048
#define HEADSC      8
#define DHC         64
#define DEPTHC      4
#define PATCH_DIMC  432
#define PATCH_KPAD  448                     // 432 rounded up to 32
#define NBATCH      32
#define MROWS       (NBATCH * SEQ)          // 25088 = 196 * 128
#define NEGBIAS     (-1.0e9f)

typedef __attribute__((ext_vector_type(16))) __bf16 v16bf;
typedef __attribute__((ext_vector_type(8)))  float  v8f;

union Frag16 { uint4 u[2]; v16bf v; };
union V8B { uint4 u; __hip_bfloat16 h[8]; };

__device__ __forceinline__ v16bf load_frag(const void* p0, const void* p1) {
    Frag16 f;
    f.u[0] = *(const uint4*)p0;
    f.u[1] = *(const uint4*)p1;
    return f.v;
}

__device__ __forceinline__ v8f wmma_bf16(v16bf a, v16bf b, v8f c) {
    return __builtin_amdgcn_wmma_f32_16x16x32_bf16(false, a, false, b, (short)0, c, false, false);
}

// ---------------- patch: fp32 [rows][K] -> bf16 [rows][Kpad] zero-padded ----------------
__global__ void cvt_pad_kernel(const float* __restrict__ src,
                               __hip_bfloat16* __restrict__ dst,
                               long long rows, int K, int Kpad) {
    long long i = (long long)blockIdx.x * blockDim.x + threadIdx.x;
    const long long total = rows * Kpad;
    const long long stride = (long long)gridDim.x * blockDim.x;
    for (; i < total; i += stride) {
        const long long r = i / Kpad;
        const int k = (int)(i - r * Kpad);
        dst[i] = (k < K) ? __float2bfloat16(src[r * K + k]) : __float2bfloat16(0.f);
    }
}

// ------- weight transpose: fp32 [K][N] -> bf16 [N][Kpad] (zero-padded K), tiled -------
__launch_bounds__(256)
__global__ void wtrans_kernel(const float* __restrict__ src,
                              __hip_bfloat16* __restrict__ dst,
                              int K, int N, int Kpad) {
    __shared__ float tile[32][33];
    const int kb = blockIdx.x * 32, nb = blockIdx.y * 32;
    const int tx = threadIdx.x & 31, ty = threadIdx.x >> 5;   // 32 x 8
    #pragma unroll
    for (int j = 0; j < 32; j += 8) {
        const int k = kb + ty + j, n = nb + tx;               // n < N always (N % 32 == 0)
        tile[ty + j][tx] = (k < K) ? src[(size_t)k * N + n] : 0.f;
    }
    __syncthreads();
    #pragma unroll
    for (int j = 0; j < 32; j += 8) {
        const int n = nb + ty + j, k = kb + tx;
        if (k < Kpad)
            dst[(size_t)n * Kpad + k] = __float2bfloat16(tile[tx][ty + j]);
    }
}

// ------- episodic keep mask, expanded over (padded) sequence positions -------
// keep_seq[n][l] = keep(t = l/49) for l < SEQ, 0 for SEQ <= l < SEQPAD
__launch_bounds__(256)
__global__ void keep_kernel(const int* __restrict__ done, float* __restrict__ keep_seq) {
    __shared__ float kt[T_STEPC];
    const int n = blockIdx.x, tid = threadIdx.x;
    if (tid < T_STEPC) {
        int any = 0;
        for (int c = tid; c < T_STEPC - 1; ++c) any |= done[n * (T_STEPC - 1) + c];
        kt[tid] = any ? 0.f : 1.f;
    }
    __syncthreads();
    for (int l = tid; l < SEQPAD; l += 256)
        keep_seq[n * SEQPAD + l] = (l < SEQ) ? kt[l / NUM_PATCH] : 0.f;
}

// ---------------- LayerNorm: one wave per row of 512 ----------------
__launch_bounds__(256)
__global__ void ln_kernel(const float* __restrict__ x,
                          const float* __restrict__ g, const float* __restrict__ b,
                          __hip_bfloat16* __restrict__ hout,  // bf16 out (or null)
                          float* __restrict__ fout,           // f32 out (if hout null)
                          int M) {
    const int lane = threadIdx.x & 31, wave = threadIdx.x >> 5;
    const int row = blockIdx.x * 8 + wave;
    if (row >= M) return;
    const float* xr = x + (size_t)row * DIMC;
    float vals[16];
    #pragma unroll
    for (int j = 0; j < 4; ++j) {
        float4 t = *(const float4*)(xr + lane * 16 + j * 4);
        vals[j*4+0] = t.x; vals[j*4+1] = t.y; vals[j*4+2] = t.z; vals[j*4+3] = t.w;
    }
    float sum = 0.f, sq = 0.f;
    #pragma unroll
    for (int j = 0; j < 16; ++j) { sum += vals[j]; sq += vals[j] * vals[j]; }
    #pragma unroll
    for (int m = 1; m < 32; m <<= 1) {
        sum += __shfl_xor(sum, m, 32);
        sq  += __shfl_xor(sq,  m, 32);
    }
    const float mean = sum * (1.f / DIMC);
    const float var  = sq * (1.f / DIMC) - mean * mean;
    const float rstd = rsqrtf(var + 1e-5f);
    #pragma unroll
    for (int j = 0; j < 16; ++j) {
        const int c = lane * 16 + j;
        const float y = (vals[j] - mean) * rstd * g[c] + b[c];
        if (hout) hout[(size_t)row * DIMC + c] = __float2bfloat16(y);
        else      fout[(size_t)row * DIMC + c] = y;
    }
}

// ---------------- generic bf16 GEMM, B pre-transposed ----------------
// C[M,N] = A[M,K] @ Bt[N,K]^T.  M%128==0, N%128==0, K%32==0 (pre-padded).
enum { EPI_EMBED = 0, EPI_BIAS_BF16 = 1, EPI_RESID = 2, EPI_GELU_BF16 = 3 };

template <int EPI>
__launch_bounds__(256)
__global__ void gemm_bf16_kernel(const __hip_bfloat16* __restrict__ A,
                                 const __hip_bfloat16* __restrict__ Bt,
                                 const float* __restrict__ bias,
                                 float* __restrict__ Cf,
                                 __hip_bfloat16* __restrict__ Cb,
                                 const float* __restrict__ aux0,   // spatial_embed
                                 const float* __restrict__ aux1,   // temporal_embed
                                 int M, int N, int K) {
    __shared__ __hip_bfloat16 As[128 * 48];   // [row][k], stride 48 (96B)
    __shared__ __hip_bfloat16 Bs[128 * 48];   // [n][k]
    const int tid  = threadIdx.x;
    const int lane = tid & 31;
    const int wave = tid >> 5;
    const int wm = wave & 3;       // 4 waves along M
    const int wn = wave >> 2;      // 2 waves along N
    const int bm = blockIdx.y * 128;
    const int bn = blockIdx.x * 128;

    // staging: thread -> (row tid>>1, 16-elem chunk (tid&1)*16); fully in-bounds.
    const int srow = tid >> 1, skc = (tid & 1) * 16;
    const __hip_bfloat16* aptr = A  + (size_t)(bm + srow) * K + skc;
    const __hip_bfloat16* bptr = Bt + (size_t)(bn + srow) * K + skc;

    uint4 ar0 = *(const uint4*)aptr, ar1 = *(const uint4*)(aptr + 8);
    uint4 br0 = *(const uint4*)bptr, br1 = *(const uint4*)(bptr + 8);

    v8f acc[2][4];
    const v8f vzero = {0.f,0.f,0.f,0.f,0.f,0.f,0.f,0.f};
    #pragma unroll
    for (int i = 0; i < 2; ++i)
        #pragma unroll
        for (int j = 0; j < 4; ++j) acc[i][j] = vzero;

    const int aoff0 = (lane < 16) ? 0 : 8;
    const int aoff1 = (lane < 16) ? 16 : 24;
    const int boff  = (lane < 16) ? 0 : 8;

    const int nK = K >> 5;
    for (int ks = 0; ks < nK; ++ks) {
        __syncthreads();
        *(uint4*)&As[srow * 48 + skc]     = ar0;
        *(uint4*)&As[srow * 48 + skc + 8] = ar1;
        *(uint4*)&Bs[srow * 48 + skc]     = br0;
        *(uint4*)&Bs[srow * 48 + skc + 8] = br1;
        __syncthreads();
        if (ks + 1 < nK) {                 // overlap next-tile loads with WMMA work
            aptr += 32; bptr += 32;
            ar0 = *(const uint4*)aptr; ar1 = *(const uint4*)(aptr + 8);
            br0 = *(const uint4*)bptr; br1 = *(const uint4*)(bptr + 8);
            if (ks + 2 < nK) {             // prefetch two tiles ahead
                __builtin_prefetch(aptr + 32, 0, 0);
                __builtin_prefetch(bptr + 32, 0, 0);
            }
        }
        v16bf af[2];
        #pragma unroll
        for (int mt = 0; mt < 2; ++mt) {
            const int r = wm * 32 + mt * 16 + (lane & 15);
            af[mt] = load_frag(&As[r * 48 + aoff0], &As[r * 48 + aoff1]);
        }
        #pragma unroll
        for (int nt = 0; nt < 4; ++nt) {
            const int c = wn * 64 + nt * 16 + (lane & 15);
            v16bf bf = load_frag(&Bs[c * 48 + boff], &Bs[c * 48 + boff + 16]);
            #pragma unroll
            for (int mt = 0; mt < 2; ++mt)
                acc[mt][nt] = wmma_bf16(af[mt], bf, acc[mt][nt]);
        }
    }

    // epilogue: D layout -> row = base + r (+8 upper half), col = lane&15
    #pragma unroll
    for (int mt = 0; mt < 2; ++mt) {
        const int rbase = bm + wm * 32 + mt * 16 + ((lane < 16) ? 0 : 8);
        #pragma unroll
        for (int nt = 0; nt < 4; ++nt) {
            const int col = bn + wn * 64 + nt * 16 + (lane & 15);
            const float bval = bias[col];
            #pragma unroll
            for (int r = 0; r < 8; ++r) {
                const int row = rbase + r;
                const float v = acc[mt][nt][r] + bval;
                if (EPI == EPI_EMBED) {
                    const int l = row % SEQ, t = l / NUM_PATCH, p = l % NUM_PATCH;
                    Cf[(size_t)row * N + col] =
                        v + aux0[p * DIMC + col] + aux1[t * DIMC + col];
                } else if (EPI == EPI_BIAS_BF16) {
                    Cb[(size_t)row * N + col] = __float2bfloat16(v);
                } else if (EPI == EPI_RESID) {
                    Cf[(size_t)row * N + col] += v;
                } else { // EPI_GELU_BF16 (exact gelu)
                    const float g = 0.5f * v * (1.f + erff(v * 0.70710678118654752f));
                    Cb[(size_t)row * N + col] = __float2bfloat16(g);
                }
            }
        }
    }
}

// ---------------- flash attention: block=(qblock, head, batch), 8 waves x 16 q rows ----
__launch_bounds__(256)
__global__ void attn_kernel(const __hip_bfloat16* __restrict__ qkv,      // [B*SEQ][1536]
                            const float* __restrict__ keep_seq,          // [B][SEQPAD]
                            __hip_bfloat16* __restrict__ obuf) {         // [B*SEQ][512]
    __shared__ __hip_bfloat16 Ks[32 * 72];       // [key][dh] stride 72
    __shared__ __hip_bfloat16 Vt[64 * 48];       // [dh][key] stride 48
    __shared__ __hip_bfloat16 Ps[8 * 16 * 48];   // per-wave P tile [row][key]
    const int tid = threadIdx.x, lane = tid & 31, wave = tid >> 5;
    const int qb = blockIdx.x, head = blockIdx.y, batch = blockIdx.z;
    const int qbase = qb * 128 + wave * 16;
    const size_t rs = 3 * DIMC;
    const __hip_bfloat16* qkvb = qkv + (size_t)batch * SEQ * rs;
    const float* keepn = keep_seq + batch * SEQPAD;
    const int half8 = (lane < 16) ? 0 : 8;

    // Q fragments (A operand), K-dim = dh split into two 32-chunks
    v16bf qf[2];
    {
        int qr = qbase + (lane & 15); if (qr >= SEQ) qr = SEQ - 1;
        const __hip_bfloat16* qp = qkvb + (size_t)qr * rs + head * DHC;
        #pragma unroll
        for (int f = 0; f < 2; ++f) {
            const int kb0 = f * 32 + ((lane < 16) ? 0 : 8);
            const int kb1 = f * 32 + ((lane < 16) ? 16 : 24);
            qf[f] = load_frag(qp + kb0, qp + kb1);
        }
    }
    float keepq[8];
    #pragma unroll
    for (int r = 0; r < 8; ++r) {
        const int q = qbase + r + half8;
        const int qc = (q < SEQ) ? q : (SEQ - 1);
        const float kq = keepn[qc];              // unconditional load, clamped index
        keepq[r] = (q < SEQ) ? kq : 1.f;
    }

    const v8f vzero = {0.f,0.f,0.f,0.f,0.f,0.f,0.f,0.f};
    v8f o[4];
    #pragma unroll
    for (int i = 0; i < 4; ++i) o[i] = vzero;
    float mst[8], lst[8];
    #pragma unroll
    for (int r = 0; r < 8; ++r) { mst[r] = -3.0e38f; lst[r] = 0.f; }
    const float scale = 0.125f;   // 64^-0.5

    // staging coords: key = tid>>3, dh chunk = (tid&7)*8
    const int skey = tid >> 3, sdh = (tid & 7) * 8;
    const __hip_bfloat16* kbase0 = qkvb + DIMC     + head * DHC + sdh;
    const __hip_bfloat16* vbase0 = qkvb + 2 * DIMC + head * DHC + sdh;

    for (int kb = 0; kb < SEQ; kb += 32) {
        __syncthreads();
        {
            const int gk = kb + skey;
            uint4 z = {0u,0u,0u,0u};
            uint4 dk = z; V8B dv; dv.u = z;
            if (gk < SEQ) {
                dk   = *(const uint4*)(kbase0 + (size_t)gk * rs);
                dv.u = *(const uint4*)(vbase0 + (size_t)gk * rs);
                if (gk + 32 < SEQ) {
                    __builtin_prefetch(kbase0 + (size_t)(gk + 32) * rs, 0, 0);
                    __builtin_prefetch(vbase0 + (size_t)(gk + 32) * rs, 0, 0);
                }
            }
            *(uint4*)&Ks[skey * 72 + sdh] = dk;
            #pragma unroll
            for (int j = 0; j < 8; ++j) Vt[(sdh + j) * 48 + skey] = dv.h[j];
        }
        __syncthreads();

        // ---- S = Q @ K^T (two 16x16 tiles over 32 keys) ----
        v8f s0v, s1v;
        {
            const int key0 = (lane & 15), key1 = 16 + (lane & 15);
            const int kkA = (lane < 16) ? 0 : 8;
            v16bf b00 = load_frag(&Ks[key0 * 72 + kkA],      &Ks[key0 * 72 + kkA + 16]);
            v16bf b01 = load_frag(&Ks[key0 * 72 + 32 + kkA], &Ks[key0 * 72 + 32 + kkA + 16]);
            v16bf b10 = load_frag(&Ks[key1 * 72 + kkA],      &Ks[key1 * 72 + kkA + 16]);
            v16bf b11 = load_frag(&Ks[key1 * 72 + 32 + kkA], &Ks[key1 * 72 + 32 + kkA + 16]);
            s0v = wmma_bf16(qf[1], b01, wmma_bf16(qf[0], b00, vzero));
            s1v = wmma_bf16(qf[1], b11, wmma_bf16(qf[0], b10, vzero));
        }

        // ---- mask bias + online softmax (fully unrolled: no indexed reg arrays) ----
        const int k0i = kb + (lane & 15), k1i = k0i + 16;
        const float kk0 = keepn[k0i];            // SEQPAD-padded: always in bounds
        const float kk1 = keepn[k1i];
        const bool v0 = (k0i < SEQ), v1 = (k1i < SEQ);
        #pragma unroll
        for (int r = 0; r < 8; ++r) {
            const float b0 = v0 ? ((keepq[r] * kk0 > 0.5f) ? 0.f : NEGBIAS) : -2.0e9f;
            const float b1 = v1 ? ((keepq[r] * kk1 > 0.5f) ? 0.f : NEGBIAS) : -2.0e9f;
            const float s0 = s0v[r] * scale + b0;
            const float s1 = s1v[r] * scale + b1;
            float m = fmaxf(s0, s1);
            #pragma unroll
            for (int msk = 1; msk < 16; msk <<= 1) m = fmaxf(m, __shfl_xor(m, msk, 32));
            const float mnew  = fmaxf(mst[r], m);
            const float alpha = __expf(mst[r] - mnew);
            const float p0 = __expf(s0 - mnew), p1 = __expf(s1 - mnew);
            float ps = p0 + p1;
            #pragma unroll
            for (int msk = 1; msk < 16; msk <<= 1) ps += __shfl_xor(ps, msk, 32);
            lst[r] = lst[r] * alpha + ps;
            mst[r] = mnew;
            #pragma unroll
            for (int ot = 0; ot < 4; ++ot) o[ot][r] *= alpha;
            // write P in A layout: [row][key]  (per-wave region; DS ops in-order per wave)
            const int prow = r + half8;
            __hip_bfloat16* pp = &Ps[(wave * 16 + prow) * 48];
            pp[(lane & 15)]      = __float2bfloat16(p0);
            pp[16 + (lane & 15)] = __float2bfloat16(p1);
        }

        // ---- O += P @ V ----
        v16bf pa;
        {
            const __hip_bfloat16* pr = &Ps[(wave * 16 + (lane & 15)) * 48];
            pa = load_frag(pr + ((lane < 16) ? 0 : 8), pr + ((lane < 16) ? 16 : 24));
        }
        #pragma unroll
        for (int ot = 0; ot < 4; ++ot) {
            const int cc = ot * 16 + (lane & 15);
            const int kf = (lane < 16) ? 0 : 8;
            v16bf vb = load_frag(&Vt[cc * 48 + kf], &Vt[cc * 48 + kf + 16]);
            o[ot] = wmma_bf16(pa, vb, o[ot]);
        }
    }

    // ---- normalize + store bf16 ----
    float linv[8];
    #pragma unroll
    for (int r = 0; r < 8; ++r) linv[r] = (lst[r] > 0.f) ? 1.f / lst[r] : 0.f;
    #pragma unroll
    for (int ot = 0; ot < 4; ++ot) {
        const int col = head * DHC + ot * 16 + (lane & 15);
        #pragma unroll
        for (int r = 0; r < 8; ++r) {
            const int q = qbase + r + half8;
            if (q < SEQ)
                obuf[((size_t)batch * SEQ + q) * DIMC + col] =
                    __float2bfloat16(o[ot][r] * linv[r]);
        }
    }
}

// ---------------- host launcher ----------------
extern "C" void kernel_launch(void* const* d_in, const int* in_sizes, int n_in,
                              void* d_out, int out_size, void* d_ws, size_t ws_size,
                              hipStream_t stream) {
    const float* patch    = (const float*)d_in[0];
    const int*   done     = (const int*)  d_in[1];
    const float* W_embed  = (const float*)d_in[2];
    const float* b_embed  = (const float*)d_in[3];
    const float* spatial  = (const float*)d_in[4];
    const float* temporal = (const float*)d_in[5];
    const float* ln1_g    = (const float*)d_in[6];
    const float* ln1_b    = (const float*)d_in[7];
    const float* Wqkv     = (const float*)d_in[8];
    const float* bqkv     = (const float*)d_in[9];
    const float* Wo       = (const float*)d_in[10];
    const float* bo       = (const float*)d_in[11];
    const float* ln2_g    = (const float*)d_in[12];
    const float* ln2_b    = (const float*)d_in[13];
    const float* W1       = (const float*)d_in[14];
    const float* b1       = (const float*)d_in[15];
    const float* W2       = (const float*)d_in[16];
    const float* b2       = (const float*)d_in[17];
    const float* out_g    = (const float*)d_in[18];
    const float* out_b    = (const float*)d_in[19];
    (void)ws_size; (void)n_in; (void)in_sizes; (void)out_size;

    // workspace carve-up (~330 MB total)
    char* base = (char*)d_ws;
    size_t off = 0;
    auto carve = [&](size_t bytes) -> char* {
        char* p = base + off;
        off = (off + bytes + 255) & ~(size_t)255;
        return p;
    };
    float*          x        = (float*)carve((size_t)MROWS * DIMC * 4);
    __hip_bfloat16* h        = (__hip_bfloat16*)carve((size_t)MROWS * DIMC * 2);
    __hip_bfloat16* qkvb     = (__hip_bfloat16*)carve((size_t)MROWS * 3 * DIMC * 2);
    __hip_bfloat16* ob       = (__hip_bfloat16*)carve((size_t)MROWS * DIMC * 2);
    __hip_bfloat16* ub       = (__hip_bfloat16*)carve((size_t)MROWS * MLP_DIMC * 2);
    float*          keepb    = (float*)carve((size_t)NBATCH * SEQPAD * 4);
    __hip_bfloat16* patch_bf = (__hip_bfloat16*)carve((size_t)MROWS * PATCH_KPAD * 2);
    // transposed bf16 weights: [N][Kpad]
    __hip_bfloat16* we_t   = (__hip_bfloat16*)carve((size_t)DIMC * PATCH_KPAD * 2);
    __hip_bfloat16* wqkv_t = (__hip_bfloat16*)carve((size_t)DEPTHC * 3 * DIMC * DIMC * 2);
    __hip_bfloat16* wo_t   = (__hip_bfloat16*)carve((size_t)DEPTHC * DIMC * DIMC * 2);
    __hip_bfloat16* w1_t   = (__hip_bfloat16*)carve((size_t)DEPTHC * MLP_DIMC * DIMC * 2);
    __hip_bfloat16* w2_t   = (__hip_bfloat16*)carve((size_t)DEPTHC * DIMC * MLP_DIMC * 2);

    // 1) setup: pad/convert patch, transpose+convert weights, keep mask
    cvt_pad_kernel<<<2048, 256, 0, stream>>>(patch, patch_bf,
                                             (long long)MROWS, PATCH_DIMC, PATCH_KPAD);
    wtrans_kernel<<<dim3(PATCH_KPAD / 32, DIMC / 32), 256, 0, stream>>>(
        W_embed, we_t, PATCH_DIMC, DIMC, PATCH_KPAD);
    for (int i = 0; i < DEPTHC; ++i) {
        wtrans_kernel<<<dim3(DIMC / 32, 3 * DIMC / 32), 256, 0, stream>>>(
            Wqkv + (size_t)i * DIMC * 3 * DIMC, wqkv_t + (size_t)i * 3 * DIMC * DIMC,
            DIMC, 3 * DIMC, DIMC);
        wtrans_kernel<<<dim3(DIMC / 32, DIMC / 32), 256, 0, stream>>>(
            Wo + (size_t)i * DIMC * DIMC, wo_t + (size_t)i * DIMC * DIMC,
            DIMC, DIMC, DIMC);
        wtrans_kernel<<<dim3(DIMC / 32, MLP_DIMC / 32), 256, 0, stream>>>(
            W1 + (size_t)i * DIMC * MLP_DIMC, w1_t + (size_t)i * MLP_DIMC * DIMC,
            DIMC, MLP_DIMC, DIMC);
        wtrans_kernel<<<dim3(MLP_DIMC / 32, DIMC / 32), 256, 0, stream>>>(
            W2 + (size_t)i * MLP_DIMC * DIMC, w2_t + (size_t)i * DIMC * MLP_DIMC,
            MLP_DIMC, DIMC, MLP_DIMC);
    }
    keep_kernel<<<NBATCH, 256, 0, stream>>>(done, keepb);

    const int mblk = MROWS / 128;  // 196
    // 2) patch embed + pos embed -> x (fp32)
    gemm_bf16_kernel<EPI_EMBED><<<dim3(DIMC / 128, mblk), 256, 0, stream>>>(
        patch_bf, we_t, b_embed, x, nullptr, spatial, temporal,
        MROWS, DIMC, PATCH_KPAD);

    // 3) transformer layers
    for (int i = 0; i < DEPTHC; ++i) {
        ln_kernel<<<MROWS / 8, 256, 0, stream>>>(x, ln1_g + i * DIMC, ln1_b + i * DIMC,
                                                 h, nullptr, MROWS);
        gemm_bf16_kernel<EPI_BIAS_BF16><<<dim3(3 * DIMC / 128, mblk), 256, 0, stream>>>(
            h, wqkv_t + (size_t)i * 3 * DIMC * DIMC, bqkv + (size_t)i * 3 * DIMC,
            nullptr, qkvb, nullptr, nullptr, MROWS, 3 * DIMC, DIMC);
        attn_kernel<<<dim3((SEQ + 127) / 128, HEADSC, NBATCH), 256, 0, stream>>>(
            qkvb, keepb, ob);
        gemm_bf16_kernel<EPI_RESID><<<dim3(DIMC / 128, mblk), 256, 0, stream>>>(
            ob, wo_t + (size_t)i * DIMC * DIMC, bo + (size_t)i * DIMC,
            x, nullptr, nullptr, nullptr, MROWS, DIMC, DIMC);
        ln_kernel<<<MROWS / 8, 256, 0, stream>>>(x, ln2_g + i * DIMC, ln2_b + i * DIMC,
                                                 h, nullptr, MROWS);
        gemm_bf16_kernel<EPI_GELU_BF16><<<dim3(MLP_DIMC / 128, mblk), 256, 0, stream>>>(
            h, w1_t + (size_t)i * MLP_DIMC * DIMC, b1 + (size_t)i * MLP_DIMC,
            nullptr, ub, nullptr, nullptr, MROWS, MLP_DIMC, DIMC);
        gemm_bf16_kernel<EPI_RESID><<<dim3(DIMC / 128, mblk), 256, 0, stream>>>(
            ub, w2_t + (size_t)i * DIMC * MLP_DIMC, b2 + (size_t)i * DIMC,
            x, nullptr, nullptr, nullptr, MROWS, DIMC, MLP_DIMC);
    }

    // 4) final LayerNorm -> d_out (fp32)
    ln_kernel<<<MROWS / 8, 256, 0, stream>>>(x, out_g, out_b, nullptr, (float*)d_out, MROWS);
}
